// GumbelQuantizer_65884798320941
// MI455X (gfx1250) — matrix operand order repaired and checked by reading, not compiled, not run on hardware
//
#include <hip/hip_runtime.h>

// MI455X / gfx1250, wave32. bf16 WMMA 16x16x32, f32 accumulate.
// GEMM2 A-tile staging uses GLOBAL_LOAD_ASYNC_TO_LDS_B128 (ASYNCcnt path).

typedef __attribute__((ext_vector_type(16))) __bf16 v16bf;
typedef __attribute__((ext_vector_type(8)))  float  v8f;

union Frag {
    v16bf f;
    uint4 u[2];
};

__device__ __forceinline__ float gelu_exact(float v) {
    // torch nn.GELU default = exact erf formulation
    return 0.5f * v * (1.0f + erff(v * 0.70710678118654752f));
}

// ---------------------------------------------------------------------------
// GEMM1: H[32768,1024] = gelu(X[32768,512] @ W1[512,1024] + b1)  (bf16 out)
// block = 256 threads (8 waves). Tile: 128 rows x 256 cols, K-chunk 64.
// grid = (32768/128, 1024/256) = (256, 4)
// ---------------------------------------------------------------------------
__global__ __launch_bounds__(256) void k_gemm1(
    const float* __restrict__ x, const float* __restrict__ W1,
    const float* __restrict__ b1, __bf16* __restrict__ H)
{
    __shared__ __attribute__((aligned(16))) __bf16 sA[128 * 72]; // [row][k], stride 72
    __shared__ __attribute__((aligned(16))) __bf16 sB[256 * 72]; // [col][k] (W1 transposed)

    const int tid  = threadIdx.x;
    const int wid  = tid >> 5;
    const int lane = tid & 31;
    const int ln   = lane & 15;
    const int half = lane >> 4;
    const int rowBase = blockIdx.x * 128;
    const int nBase   = blockIdx.y * 256;
    const int waveM = wid >> 2;  // 0..1
    const int waveN = wid & 3;   // 0..3

    v8f c[4][4];
#pragma unroll
    for (int i = 0; i < 4; ++i)
#pragma unroll
        for (int j = 0; j < 4; ++j) c[i][j] = (v8f)(0.0f);

    for (int kk = 0; kk < 512; kk += 64) {
        __syncthreads();
        // stage A: X tile 128x64 f32 -> bf16 (row-major in LDS)
#pragma unroll
        for (int i = 0; i < 8; ++i) {
            int q  = tid + 256 * i;
            int r  = q >> 4;
            int c0 = (q & 15) << 2;
            const float4 v = *(const float4*)(x + (size_t)(rowBase + r) * 512 + kk + c0);
            __bf16* d = &sA[r * 72 + c0];
            d[0] = (__bf16)v.x; d[1] = (__bf16)v.y; d[2] = (__bf16)v.z; d[3] = (__bf16)v.w;
        }
        // stage B: W1 tile 64x256 f32 -> bf16, transposed to [col][k]
#pragma unroll
        for (int i = 0; i < 16; ++i) {
            int q  = tid + 256 * i;
            int k  = q >> 6;
            int n0 = (q & 63) << 2;
            const float4 v = *(const float4*)(W1 + (size_t)(kk + k) * 1024 + nBase + n0);
            sB[(n0 + 0) * 72 + k] = (__bf16)v.x;
            sB[(n0 + 1) * 72 + k] = (__bf16)v.y;
            sB[(n0 + 2) * 72 + k] = (__bf16)v.z;
            sB[(n0 + 3) * 72 + k] = (__bf16)v.w;
        }
        if (kk + 64 < 512) { // prefetch next weight chunk into GL2
            __builtin_prefetch(W1 + (size_t)(kk + 64 + (tid & 63)) * 1024 + nBase, 0, 1);
        }
        __syncthreads();

#pragma unroll
        for (int ks = 0; ks < 64; ks += 32) {
            Frag a[4], b[4];
            const int koffA = ks + half * 8;   // A 16x32 bf16 lane layout
            const int koffB = ks + half * 8;   // B 32x16 bf16 lane layout
#pragma unroll
            for (int im = 0; im < 4; ++im) {
                int row = waveM * 64 + im * 16 + ln;
                a[im].u[0] = *(const uint4*)&sA[row * 72 + koffA];
                a[im].u[1] = *(const uint4*)&sA[row * 72 + koffA + 16];
            }
#pragma unroll
            for (int in = 0; in < 4; ++in) {
                int col = waveN * 64 + in * 16 + ln;
                b[in].u[0] = *(const uint4*)&sB[col * 72 + koffB];
                b[in].u[1] = *(const uint4*)&sB[col * 72 + koffB + 16];
            }
#pragma unroll
            for (int im = 0; im < 4; ++im)
#pragma unroll
                for (int in = 0; in < 4; ++in)
                    c[im][in] = __builtin_amdgcn_wmma_f32_16x16x32_bf16(
                        false, a[im].f, false, b[in].f, (short)0, c[im][in], false, false);
        }
    }

    // epilogue: + b1, exact GELU, store bf16
#pragma unroll
    for (int in = 0; in < 4; ++in) {
        int ncol = nBase + waveN * 64 + in * 16 + ln;
        float bias = b1[ncol];
#pragma unroll
        for (int im = 0; im < 4; ++im) {
            int m0 = rowBase + waveM * 64 + im * 16 + half * 8;
#pragma unroll
            for (int r = 0; r < 8; ++r) {
                float v = c[im][in][r] + bias;
                H[(size_t)(m0 + r) * 1024 + ncol] = (__bf16)gelu_exact(v);
            }
        }
    }
}

// ---------------------------------------------------------------------------
// GEMM2 + fused epilogue:
//   logits[64,640] = H_tile @ W2 + b2 + gumbels; per (row,group) argmax over
//   320 cols; out[row, g*128:+128] = emb[idx].
// A-tile (already bf16) staged with GLOBAL_LOAD_ASYNC_TO_LDS_B128.
// block = 256 threads (8 waves), wave w owns cols [80w, 80w+80). grid = 512.
// ---------------------------------------------------------------------------
__global__ __launch_bounds__(256) void k_gemm2(
    const __bf16* __restrict__ H, const float* __restrict__ W2,
    const float* __restrict__ b2, const float* __restrict__ gumbels,
    const float* __restrict__ emb, float* __restrict__ out)
{
    __shared__ __attribute__((aligned(16))) __bf16 sA[64 * 40];   // [row][k]
    __shared__ __attribute__((aligned(16))) __bf16 sB[640 * 40];  // [col][k]
    __shared__ float pVal[64][8];
    __shared__ int   pIdx[64][8];
    __shared__ int   winIdx[64][2];

    const int tid  = threadIdx.x;
    const int wid  = tid >> 5;
    const int lane = tid & 31;
    const int ln   = lane & 15;
    const int half = lane >> 4;
    const int rowBase = blockIdx.x * 64;
    const int colBase = wid * 80;

    // per-thread source/dest for the async A-tile copy (one b128 per thread)
    const int ar  = tid >> 2;           // row 0..63
    const int ac0 = (tid & 3) << 3;     // k offset 0/8/16/24
    const __bf16* aSrcBase = H + (size_t)(rowBase + ar) * 1024 + ac0;
    const unsigned aLds = (unsigned)(uintptr_t)&sA[ar * 40 + ac0]; // LDS = addr[31:0]

    v8f c[4][5];
#pragma unroll
    for (int i = 0; i < 4; ++i)
#pragma unroll
        for (int j = 0; j < 5; ++j) c[i][j] = (v8f)(0.0f);

    for (int kk = 0; kk < 1024; kk += 32) {
        __syncthreads();
        // stage A: H tile 64x32 bf16, async DMA global -> LDS (ASYNCcnt path)
        {
            const __bf16* gp = aSrcBase + kk;
            asm volatile("global_load_async_to_lds_b128 %0, %1, off"
                         :: "v"(aLds), "v"(gp)
                         : "memory");
        }
        // stage B: W2 tile 32x640 f32 -> bf16 transposed to [col][k]
#pragma unroll
        for (int i = 0; i < 20; ++i) {
            int q  = tid + 256 * i;
            int k  = q / 160;
            int n0 = (q % 160) << 2;
            const float4 v = *(const float4*)(W2 + (size_t)(kk + k) * 640 + n0);
            sB[(n0 + 0) * 40 + k] = (__bf16)v.x;
            sB[(n0 + 1) * 40 + k] = (__bf16)v.y;
            sB[(n0 + 2) * 40 + k] = (__bf16)v.z;
            sB[(n0 + 3) * 40 + k] = (__bf16)v.w;
        }
        if (kk + 32 < 1024) {
            __builtin_prefetch(W2 + (size_t)(kk + 32 + (tid & 31)) * 640, 0, 1);
        }
        // drain async copies before publishing the tile
        asm volatile("s_wait_asynccnt 0x0" ::: "memory");
        __syncthreads();

        Frag a[4], b[5];
        const int koffA = half * 8;
        const int koffB = half * 8;
#pragma unroll
        for (int im = 0; im < 4; ++im) {
            int row = im * 16 + ln;
            a[im].u[0] = *(const uint4*)&sA[row * 40 + koffA];
            a[im].u[1] = *(const uint4*)&sA[row * 40 + koffA + 16];
        }
#pragma unroll
        for (int ct = 0; ct < 5; ++ct) {
            int col = colBase + ct * 16 + ln;
            b[ct].u[0] = *(const uint4*)&sB[col * 40 + koffB];
            b[ct].u[1] = *(const uint4*)&sB[col * 40 + koffB + 16];
        }
#pragma unroll
        for (int im = 0; im < 4; ++im)
#pragma unroll
            for (int ct = 0; ct < 5; ++ct)
                c[im][ct] = __builtin_amdgcn_wmma_f32_16x16x32_bf16(
                    false, a[im].f, false, b[ct].f, (short)0, c[im][ct], false, false);
    }

    // epilogue 1: + b2 + gumbels, per-row argmax within this wave's 80 cols
    const int g = wid >> 2;              // waves 0-3 -> group 0, 4-7 -> group 1
    const float NEG = -3.402823466e38f;
#pragma unroll
    for (int im = 0; im < 4; ++im) {
#pragma unroll
        for (int r = 0; r < 8; ++r) {
            int mloc  = im * 16 + half * 8 + r;
            int token = rowBase + mloc;
            float best = NEG;
            int bestc  = 0x7fffffff;
#pragma unroll
            for (int ct = 0; ct < 5; ++ct) {
                int ncol = colBase + ct * 16 + ln;
                float v = c[im][ct][r] + b2[ncol]
                        + gumbels[((size_t)token * 2 + g) * 320 + (ncol - g * 320)];
                if (v > best) { best = v; bestc = ncol; }
            }
            // butterfly max over the 16 lanes of this half (wave32)
#pragma unroll
            for (int m = 1; m < 16; m <<= 1) {
                float ov = __shfl_xor(best, m, 32);
                int   oc = __shfl_xor(bestc, m, 32);
                if (ov > best || (ov == best && oc < bestc)) { best = ov; bestc = oc; }
            }
            if (ln == 0) { pVal[mloc][wid] = best; pIdx[mloc][wid] = bestc; }
        }
    }
    __syncthreads();

    // epilogue 2: reduce 4 wave-partials per (row, group) -> code index
    if (tid < 128) {
        int row = tid & 63;
        int gg  = tid >> 6;
        float best = NEG;
        int bestc  = 0x7fffffff;
#pragma unroll
        for (int w = 0; w < 4; ++w) {
            float v = pVal[row][gg * 4 + w];
            int  ci = pIdx[row][gg * 4 + w];
            if (v > best || (v == best && ci < bestc)) { best = v; bestc = ci; }
        }
        winIdx[row][gg] = bestc - gg * 320;  // code in [0,320)
    }
    __syncthreads();

    // epilogue 3: gather emb[code] (128 f32) -> out; 2 threads per (row,group)
    {
        int p   = tid >> 1;
        int hc  = tid & 1;
        int row = p & 63;
        int gg  = p >> 6;
        int code = winIdx[row][gg];
        const float* src = emb + (size_t)code * 128 + hc * 64;
        float* dst = out + (size_t)(rowBase + row) * 256 + gg * 128 + hc * 64;
#pragma unroll
        for (int j = 0; j < 16; ++j)
            *(float4*)(dst + 4 * j) = *(const float4*)(src + 4 * j);
    }
}

extern "C" void kernel_launch(void* const* d_in, const int* in_sizes, int n_in,
                              void* d_out, int out_size, void* d_ws, size_t ws_size,
                              hipStream_t stream) {
    const float* x       = (const float*)d_in[0];
    const float* gumbels = (const float*)d_in[1];
    const float* W1      = (const float*)d_in[2];
    const float* b1      = (const float*)d_in[3];
    const float* W2      = (const float*)d_in[4];
    const float* b2      = (const float*)d_in[5];
    const float* emb     = (const float*)d_in[6];
    float* out = (float*)d_out;

    __bf16* H = (__bf16*)d_ws;  // 32768 x 1024 bf16 = 64 MB intermediate

    dim3 g1(256, 4);
    k_gemm1<<<g1, 256, 0, stream>>>(x, W1, b1, H);
    k_gemm2<<<512, 256, 0, stream>>>(H, W2, b2, gumbels, emb, out);
}
